// GroupTorchGRU_48318382080192
// MI455X (gfx1250) — matrix operand order repaired and checked by reading, not compiled, not run on hardware
//
#include <hip/hip_runtime.h>

// GroupTorchGRU on gfx1250: bf16 WMMA (f32 accumulate) GEMMs + fused GRU epilogue.
// B=8192, U=16, I=H=128.  Fused K = I+H = 256 for r/z gates; n-gate split into
// xn (K 0..127, inputs*w_ih) and hn (K 128..255, hidden*w_hh) accumulators.

typedef __attribute__((ext_vector_type(2)))  unsigned int   uintx2;
typedef __attribute__((ext_vector_type(8)))  unsigned short ushortx8;
typedef __attribute__((ext_vector_type(16))) unsigned short ushortx16;
typedef __attribute__((ext_vector_type(2)))  __bf16         bfx2;
typedef __attribute__((ext_vector_type(16))) __bf16         bfx16;
typedef __attribute__((ext_vector_type(8)))  float          fx8;

#define U_TOT 16
#define HDIM  128
#define G3    384
#define LDA   136   // padded bf16 row stride: 272 bytes (16B aligned, bank step 4)

// f32 -> packed bf16 pair (RNE).  Plain casts let the backend pick the
// hardware cvt (v_cvt_pk_bf16_f32) instead of a bfe/add3 bit sequence.
__device__ __forceinline__ unsigned int pk_bf16(float a, float b) {
  bfx2 p;
  p[0] = (__bf16)a;
  p[1] = (__bf16)b;
  return __builtin_bit_cast(unsigned int, p);
}

__device__ __forceinline__ bfx16 make_frag(const unsigned short* p0,
                                           const unsigned short* p1) {
  ushortx8 lo = *(const ushortx8*)p0;   // ds_load_b128
  ushortx8 hi = *(const ushortx8*)p1;   // ds_load_b128
  ushortx16 v = __builtin_shufflevector(lo, hi,
      0,1,2,3,4,5,6,7,8,9,10,11,12,13,14,15);
  return __builtin_bit_cast(bfx16, v);
}

__global__ __launch_bounds__(256)
void gru_wmma_kernel(const float* __restrict__ inputs,
                     const float* __restrict__ hidden,
                     const float* __restrict__ w_ih,
                     const float* __restrict__ w_hh,
                     const float* __restrict__ b_ih,
                     const float* __restrict__ b_hh,
                     float* __restrict__ out) {
  __shared__ unsigned short As[128 * LDA];  // 128 batch rows x 128 K (bf16)
  __shared__ unsigned short Ws[48 * LDA];   // 3 gates x 16 cols x 128 K (bf16)

  const int t      = threadIdx.x;
  const int mBlock = blockIdx.x;        // 64 blocks of 128 batch rows
  const int h0     = blockIdx.y * 16;   // 8 tiles of 16 h-columns
  const int u      = blockIdx.z;        // 16 units

  const int wv    = t >> 5;             // wave 0..7
  const int lane  = t & 31;
  const int half  = lane >> 4;          // lane-half selects K sub-slices
  const int lan16 = lane & 15;

  fx8 accR = {};      // r gate, fused K=256
  fx8 accZ = {};      // z gate, fused K=256
  fx8 accN[2] = {};   // [0]=xn: inputs*w_ih[2H:], [1]=hn: hidden*w_hh[2H:]

#pragma unroll
  for (int kc = 0; kc < 2; ++kc) {
    const float* asrc = kc ? hidden : inputs;
    const float* wsrc = kc ? w_hh   : w_ih;

    // ---- stage A tile: 128 rows x 128 cols, f32 -> bf16 (coalesced) ----
    // Each iteration: 32 consecutive lanes read one full 512B row segment.
#pragma unroll
    for (int i = 0; i < 16; ++i) {
      const int idx = i * 256 + t;          // 0 .. 4095 float4 slots
      const int r   = idx >> 5;             // batch row within tile
      const int c4  = (idx & 31) * 4;       // float column
      const int b   = mBlock * 128 + r;
      float4 v = *(const float4*)(asrc + ((size_t)(b * U_TOT + u)) * HDIM + c4);
      uintx2 pkd;
      pkd[0] = pk_bf16(v.x, v.y);
      pkd[1] = pk_bf16(v.z, v.w);
      *(uintx2*)(&As[r * LDA + c4]) = pkd;  // ds_store_b64
    }
    // ---- stage W tile: 48 gate-rows x 128 cols ----
#pragma unroll
    for (int i = 0; i < 6; ++i) {
      const int idx  = i * 256 + t;         // 0 .. 1535 float4 slots
      const int rw   = idx >> 5;            // 0..47 = gate*16 + n
      const int c4   = (idx & 31) * 4;
      const int gate = rw >> 4;
      const int n    = rw & 15;
      const int grow = gate * HDIM + h0 + n;
      float4 v = *(const float4*)(wsrc + ((size_t)(u * G3 + grow)) * HDIM + c4);
      uintx2 pkd;
      pkd[0] = pk_bf16(v.x, v.y);
      pkd[1] = pk_bf16(v.z, v.w);
      *(uintx2*)(&Ws[rw * LDA + c4]) = pkd; // ds_store_b64
    }
    __syncthreads();

    // ---- 4 K-steps of v_wmma_f32_16x16x32_bf16 per gate ----
    const unsigned short* arow = &As[(wv * 16 + lan16) * LDA];
#pragma unroll
    for (int kk = 0; kk < 4; ++kk) {
      const int k0 = kk * 32;
      // A frag: 16 consecutive K starting k0 + half*16
      bfx16 af = make_frag(arow + k0 + half * 16,
                           arow + k0 + half * 16 + 8);
      // B frags: column = lane%16, 16 consecutive K starting k0 + half*16
      const unsigned short* wr = &Ws[(0 * 16 + lan16) * LDA + k0 + half * 16];
      const unsigned short* wz = &Ws[(1 * 16 + lan16) * LDA + k0 + half * 16];
      const unsigned short* wn = &Ws[(2 * 16 + lan16) * LDA + k0 + half * 16];
      bfx16 br = make_frag(wr, wr + 8);
      bfx16 bz = make_frag(wz, wz + 8);
      bfx16 bn = make_frag(wn, wn + 8);

      accR = __builtin_amdgcn_wmma_f32_16x16x32_bf16(false, af, false, br,
                                                     (short)0, accR, false, false);
      accZ = __builtin_amdgcn_wmma_f32_16x16x32_bf16(false, af, false, bz,
                                                     (short)0, accZ, false, false);
      accN[kc] = __builtin_amdgcn_wmma_f32_16x16x32_bf16(false, af, false, bn,
                                                         (short)0, accN[kc],
                                                         false, false);
    }
    __syncthreads();
  }

  // ---- GRU epilogue: D layout -> row M = j + half*8, col N = lane%16 ----
  const int h = h0 + lan16;
  const float bir = b_ih[u * G3 + h];
  const float bhr = b_hh[u * G3 + h];
  const float biz = b_ih[u * G3 + HDIM + h];
  const float bhz = b_hh[u * G3 + HDIM + h];
  const float bin = b_ih[u * G3 + 2 * HDIM + h];
  const float bhn = b_hh[u * G3 + 2 * HDIM + h];

  const int mbase = mBlock * 128 + wv * 16 + half * 8;
#pragma unroll
  for (int j = 0; j < 8; ++j) {
    const int b = mbase + j;
    const size_t oidx = ((size_t)(b * U_TOT + u)) * HDIM + h;
    const float hid = hidden[oidx];
    const float r = 1.0f / (1.0f + __expf(-(accR[j] + bir + bhr)));
    const float z = 1.0f / (1.0f + __expf(-(accZ[j] + biz + bhz)));
    // tanh(x) = 1 - 2/(exp(2x)+1)
    const float nx = accN[0][j] + bin + r * (accN[1][j] + bhn);
    const float e2 = __expf(2.0f * nx);
    const float ng = 1.0f - 2.0f / (e2 + 1.0f);
    out[oidx] = (1.0f - z) * ng + z * hid;
  }
}

extern "C" void kernel_launch(void* const* d_in, const int* in_sizes, int n_in,
                              void* d_out, int out_size, void* d_ws, size_t ws_size,
                              hipStream_t stream) {
  const float* inputs = (const float*)d_in[0];
  const float* hidden = (const float*)d_in[1];
  const float* w_ih   = (const float*)d_in[2];
  const float* w_hh   = (const float*)d_in[3];
  const float* b_ih   = (const float*)d_in[4];
  const float* b_hh   = (const float*)d_in[5];
  float* out = (float*)d_out;

  dim3 grid(64, 8, 16);   // (batch blocks of 128, h tiles of 16, units)
  dim3 block(256);        // 8 wave32
  gru_wmma_kernel<<<grid, block, 0, stream>>>(inputs, hidden, w_ih, w_hh,
                                              b_ih, b_hh, out);
}
